// LSTM_29875792511458
// MI455X (gfx1250) — matrix operand order repaired and checked by reading, not compiled, not run on hardware
//
#include <hip/hip_runtime.h>
#include <cstdint>
#include <cstddef>

#define H 1024
#define BSZ 32
#define TT 512
#define G4 4096            // 4*H
#define KDIM 1024
#define APITCH 1032        // LDS row pitch in halves (rows shift by 4 banks)
#define NBLK 64            // scan grid size (H/16)

typedef _Float16 v16h __attribute__((ext_vector_type(16)));
typedef _Float16 v8h  __attribute__((ext_vector_type(8)));
typedef _Float16 v4h  __attribute__((ext_vector_type(4)));
typedef float    v8f  __attribute__((ext_vector_type(8)));

// A-fragment (16x32 f16): lane<16 -> row=lane, K=kb+0..15;
// lane>=16 -> row=lane-16, K=kb+16..31. p is pre-offset by kb+koffA.
__device__ __forceinline__ v16h load_a_frag(const _Float16* p) {
  v8h lo = *(const v8h*)(p);
  v8h hi = *(const v8h*)(p + 8);
  v16h r;
#pragma unroll
  for (int i = 0; i < 8; ++i) { r[i] = lo[i]; r[i + 8] = hi[i]; }
  return r;
}

__device__ __forceinline__ float sigmoidf_(float x) {
  return 1.0f / (1.0f + __expf(-x));
}

// ---------------------------------------------------------------------------
// Pack W [K=1024][N=4096] fp32 -> WT [N=4096][K=1024] f16. B-fragment becomes
// 16 contiguous halves per lane (lane<16: col=lane, K=kb+0..15; lane>=16:
// col=lane-16, K=kb+16..31).
// ---------------------------------------------------------------------------
__global__ void pack_wT_kernel(const float* __restrict__ W, _Float16* __restrict__ WT) {
  const size_t idx = (size_t)blockIdx.x * 256 + threadIdx.x;   // over 4096*1024
  const int n = (int)(idx >> 10);
  const int k = (int)(idx & 1023);
  WT[idx] = (_Float16)W[(size_t)k * G4 + n];
}

__global__ void zero_kernel(float* __restrict__ p, int n) {
  const int i = blockIdx.x * 256 + threadIdx.x;
  if (i < n) p[i] = 0.0f;
}

// ---------------------------------------------------------------------------
// x_proj GEMM: C[32 x 4096] tile = A[32 x 1024] * WT^T + bias
// grid = (4096/64, rows/32); 128 threads = 4 waves; wave owns 16 cols, 2 m-tiles.
// ---------------------------------------------------------------------------
__global__ void __launch_bounds__(128)
xproj_gemm_kernel(const float* __restrict__ A,      // [gridDim.y*32, 1024] fp32
                  const _Float16* __restrict__ BT,  // [4096, 1024] f16 (W^T)
                  const float* __restrict__ bias,   // [4096]
                  float* __restrict__ C)            // [gridDim.y*32, 4096] fp32
{
  __shared__ __align__(16) _Float16 As[32 * APITCH];
  const int tid = threadIdx.x;
  const size_t rowbase = (size_t)blockIdx.y * 32;
  const float* Ar = A + rowbase * KDIM;

  for (int i = tid; i < (32 * KDIM) / 4; i += 128) {
    const float4 v = ((const float4*)Ar)[i];
    const int k4  = (i * 4) & (KDIM - 1);
    const int row = (i * 4) >> 10;
    v4h p; p[0] = (_Float16)v.x; p[1] = (_Float16)v.y;
           p[2] = (_Float16)v.z; p[3] = (_Float16)v.w;
    *(v4h*)(&As[row * APITCH + k4]) = p;
  }
  __syncthreads();

  const int wave = tid >> 5;          // wave32
  const int lane = tid & 31;
  const int nloc = lane & 15;
  const int ncol = blockIdx.x * 64 + wave * 16 + nloc;
  const int koffA = (lane < 16) ? 0 : 16;
  const int koffB = (lane < 16) ? 0 : 16;
  const _Float16* brow = BT + (size_t)ncol * KDIM;

  v8f acc0 = {}; v8f acc1 = {};
#pragma unroll 4
  for (int kt = 0; kt < KDIM / 32; ++kt) {
    const int kb = kt * 32;
    const v16h bf = *(const v16h*)(brow + kb + koffB);
    const v16h a0 = load_a_frag(&As[nloc * APITCH + kb + koffA]);
    const v16h a1 = load_a_frag(&As[(nloc + 16) * APITCH + kb + koffA]);
    acc0 = __builtin_amdgcn_wmma_f32_16x16x32_f16(false, a0, false, bf, (short)0, acc0, false, false);
    acc1 = __builtin_amdgcn_wmma_f32_16x16x32_f16(false, a1, false, bf, (short)0, acc1, false, false);
  }

  const float bv = bias[ncol];
  const int mbase = (lane < 16) ? 0 : 8;
#pragma unroll
  for (int r = 0; r < 8; ++r) {
    C[(rowbase + mbase + r) * (size_t)G4 + ncol]      = acc0[r] + bv;
    C[(rowbase + 16 + mbase + r) * (size_t)G4 + ncol] = acc1[r] + bv;
  }
}

// ---------------------------------------------------------------------------
// Persistent recurrence: one launch runs all 512 timesteps.
// grid = 64 blocks (block nb owns hidden units [nb*16, nb*16+16) forever);
// 128 threads = 4 waves, wave g computes gate g's 32x16 tile (K=1024).
// Cell state c lives in registers for the whole scan (4 elems/thread).
// h is exchanged via global ping-pong + a device-scope counter barrier per step.
// ---------------------------------------------------------------------------
__global__ void __launch_bounds__(128)
lstm_scan_kernel(float* __restrict__ hb0,           // [32,1024] ping (zeroed = h_0)
                 float* __restrict__ hb1,           // [32,1024] pong
                 const _Float16* __restrict__ WhhT, // [4096,1024] f16
                 const float* __restrict__ xproj,   // [32*512, 4096]
                 float* __restrict__ y,             // [32, 512, 1024]
                 float* __restrict__ hn_out,        // [32,1024]
                 float* __restrict__ cn_out,        // [32,1024]
                 unsigned* __restrict__ cnt)        // [512] zeroed
{
  __shared__ __align__(16) unsigned char smem[32 * APITCH * 2]; // 66048 B
  _Float16* Hs = (_Float16*)smem;      // staged h (f16, pitched)
  float*    Gs = (float*)smem;         // gate tiles [32][65], reused after barrier

  const int tid  = threadIdx.x;
  const int gate = tid >> 5;
  const int lane = tid & 31;
  const int nloc = lane & 15;
  const int hn   = blockIdx.x * 16 + nloc;
  const int ncol = gate * H + hn;
  const int koffA = (lane < 16) ? 0 : 16;
  const int koffB = (lane < 16) ? 0 : 16;
  const _Float16* brow = WhhT + (size_t)ncol * KDIM;
  const int mbase = (lane < 16) ? 0 : 8;
  const int col = gate * 16 + nloc;

  float creg[4] = {0.0f, 0.0f, 0.0f, 0.0f};   // persistent cell state

  for (int t = 0; t < TT; ++t) {
    const float* hprev = (t & 1) ? hb1 : hb0;
    float*       hnext = (t & 1) ? hb0 : hb1;

    // stage full h_{t-1} (32x1024 fp32 -> f16 pitched LDS)
    for (int i = tid; i < (BSZ * KDIM) / 4; i += 128) {
      const float4 v = ((const float4*)hprev)[i];
      const int k4  = (i * 4) & (KDIM - 1);
      const int row = (i * 4) >> 10;
      v4h p; p[0] = (_Float16)v.x; p[1] = (_Float16)v.y;
             p[2] = (_Float16)v.z; p[3] = (_Float16)v.w;
      *(v4h*)(&Hs[row * APITCH + k4]) = p;
    }
    __syncthreads();

    v8f acc0 = {}; v8f acc1 = {};
#pragma unroll 4
    for (int kt = 0; kt < KDIM / 32; ++kt) {
      const int kb = kt * 32;
      const v16h bf = *(const v16h*)(brow + kb + koffB);
      const v16h a0 = load_a_frag(&Hs[nloc * APITCH + kb + koffA]);
      const v16h a1 = load_a_frag(&Hs[(nloc + 16) * APITCH + kb + koffA]);
      acc0 = __builtin_amdgcn_wmma_f32_16x16x32_f16(false, a0, false, bf, (short)0, acc0, false, false);
      acc1 = __builtin_amdgcn_wmma_f32_16x16x32_f16(false, a1, false, bf, (short)0, acc1, false, false);
    }

    __syncthreads();  // all Hs reads done before Gs overwrites the same LDS

#pragma unroll
    for (int r = 0; r < 8; ++r) {
      const int m0 = mbase + r, m1 = 16 + mbase + r;
      Gs[m0 * 65 + col] = acc0[r] + xproj[((size_t)m0 * TT + t) * G4 + ncol];
      Gs[m1 * 65 + col] = acc1[r] + xproj[((size_t)m1 * TT + t) * G4 + ncol];
    }
    __syncthreads();

#pragma unroll
    for (int q = 0; q < 4; ++q) {
      const int e = tid + 128 * q;            // 512 = 32 rows x 16 cols
      const int b = e >> 4, j = e & 15;
      const float gi = sigmoidf_(Gs[b * 65 + j]);
      const float gf = sigmoidf_(Gs[b * 65 + 16 + j]);
      const float gg = tanhf(Gs[b * 65 + 32 + j]);
      const float go = sigmoidf_(Gs[b * 65 + 48 + j]);
      const int n = blockIdx.x * 16 + j;
      const float c = gf * creg[q] + gi * gg;
      const float h = go * tanhf(c);
      creg[q] = c;
      hnext[b * H + n] = h;
      y[((size_t)b * TT + t) * H + n] = h;
      if (t == TT - 1) { hn_out[b * H + n] = h; cn_out[b * H + n] = c; }
    }

    // ---- device-scope grid barrier (release h writes / acquire before reads)
    __threadfence();
    __syncthreads();
    if (tid == 0) {
      __hip_atomic_fetch_add(&cnt[t], 1u, __ATOMIC_RELEASE, __HIP_MEMORY_SCOPE_AGENT);
      while (__hip_atomic_load(&cnt[t], __ATOMIC_ACQUIRE, __HIP_MEMORY_SCOPE_AGENT) < (unsigned)NBLK) {
        __builtin_amdgcn_s_sleep(1);
      }
    }
    __syncthreads();
  }
}

// ---------------------------------------------------------------------------
extern "C" void kernel_launch(void* const* d_in, const int* in_sizes, int n_in,
                              void* d_out, int out_size, void* d_ws, size_t ws_size,
                              hipStream_t stream) {
  (void)in_sizes; (void)n_in; (void)out_size; (void)ws_size;

  const float* x    = (const float*)d_in[0];
  const float* wih0 = (const float*)d_in[1];
  const float* whh0 = (const float*)d_in[2];
  const float* b0   = (const float*)d_in[3];
  const float* wih1 = (const float*)d_in[4];
  const float* whh1 = (const float*)d_in[5];
  const float* b1   = (const float*)d_in[6];
  float* out = (float*)d_out;

  char* ws = (char*)d_ws;
  const size_t WSZ = (size_t)G4 * KDIM * sizeof(_Float16);      // 8 MiB each
  _Float16* wih0T = (_Float16*)(ws + 0 * WSZ);
  _Float16* whh0T = (_Float16*)(ws + 1 * WSZ);
  _Float16* wih1T = (_Float16*)(ws + 2 * WSZ);
  _Float16* whh1T = (_Float16*)(ws + 3 * WSZ);
  float* xproj = (float*)(ws + 4 * WSZ);                        // 256 MiB
  float* y0    = (float*)(ws + 4 * WSZ + (size_t)BSZ * TT * G4 * sizeof(float)); // 64 MiB
  float* hbuf0 = y0 + (size_t)BSZ * TT * H;                     // [32,1024]
  float* hbuf1 = hbuf0 + BSZ * H;                               // [32,1024]
  unsigned* cnt = (unsigned*)(hbuf1 + BSZ * H);                 // [512]
  const int stateWords = 2 * BSZ * H + TT;                      // hbuf0,hbuf1,cnt

  const int packBlocks = (G4 * KDIM) / 256;
  pack_wT_kernel<<<packBlocks, 256, 0, stream>>>(wih0, wih0T);
  pack_wT_kernel<<<packBlocks, 256, 0, stream>>>(whh0, whh0T);
  pack_wT_kernel<<<packBlocks, 256, 0, stream>>>(wih1, wih1T);
  pack_wT_kernel<<<packBlocks, 256, 0, stream>>>(whh1, whh1T);

  float* yout[2]          = { y0, out };        // layer1 writes y1 region of out
  const _Float16* wihT[2] = { wih0T, wih1T };
  const _Float16* whhT[2] = { whh0T, whh1T };
  const float*    bias[2] = { b0, b1 };
  const float*    ain[2]  = { x, y0 };
  float* hn_dst = out + (size_t)BSZ * TT * H;   // h_n [2,32,1024]
  float* cn_dst = hn_dst + 2 * BSZ * H;         // c_n [2,32,1024]

  for (int layer = 0; layer < 2; ++layer) {
    // zero h ping-pong (h_0 = 0) and barrier counters
    zero_kernel<<<(stateWords + 255) / 256, 256, 0, stream>>>(hbuf0, stateWords);

    // x_proj = A @ w_ih + b for all timesteps (time-parallel WMMA GEMM)
    dim3 ggrid(G4 / 64, (BSZ * TT) / 32);
    xproj_gemm_kernel<<<ggrid, 128, 0, stream>>>(ain[layer], wihT[layer], bias[layer], xproj);

    // persistent 512-step recurrence, single launch
    lstm_scan_kernel<<<dim3(NBLK), 128, 0, stream>>>(
        hbuf0, hbuf1, whhT[layer], xproj, yout[layer],
        hn_dst + (size_t)layer * BSZ * H, cn_dst + (size_t)layer * BSZ * H, cnt);
  }
}
